// MultiHeadAttention_66786741452878
// MI455X (gfx1250) — compile-verified, matrix operand order checked by reading, not run on hardware
//
#include <hip/hip_runtime.h>

// Problem constants (reference: B=2, S=2048, D=1024, H=16, HD=64)
#define B_  2
#define S_  2048
#define D_  1024
#define H_  16
#define HD_ 64
#define GM  (B_ * S_)   // 4096 rows for the projections
#define GN  D_          // 1024
#define GK  D_          // 1024

typedef __attribute__((ext_vector_type(16))) __bf16 v16bf;
typedef __attribute__((ext_vector_type(8)))  __bf16 v8bf;
typedef __attribute__((ext_vector_type(8)))  float  v8f;

// ---------------------------------------------------------------------------
// WMMA helper: D = A(16x32 bf16) * B(32x16 bf16) + C(16x16 f32)
// ---------------------------------------------------------------------------
__device__ __forceinline__ v8f wmma_bf16(v16bf a, v16bf b, v8f c) {
  return __builtin_amdgcn_wmma_f32_16x16x32_bf16(
      /*neg_a=*/false, a, /*neg_b=*/false, b,
      /*c_mod=*/(short)0, c, /*reuse_a=*/false, /*reuse_b=*/false);
}

// ---------------------------------------------------------------------------
// Fragment loaders.
// ISA 7.12.2 (16-bit A 16x32): lane L holds row M = L%16; with half = L/16 its
// 16 elements are the two contiguous K-chunks [8*half, +8) and [16+8*half, +8).
// For Y = X * W^T with X row-major [M,K] and W row-major [N,K], the B-operand
// (K x N) gathers W row N = L%16 with the same two chunks -> identical load
// pattern for both operands, two global_load_b128 each, no transpose needed.
// ---------------------------------------------------------------------------
__device__ __forceinline__ v16bf load_frag_bf16(const __bf16* base, int row_stride) {
  int lane = threadIdx.x & 31;
  int r    = lane & 15;
  int half = lane >> 4;
  const __bf16* p = base + (size_t)r * row_stride + half * 8;
  v8bf lo = *(const v8bf*)(p);        // K = 8*half .. +8
  v8bf hi = *(const v8bf*)(p + 16);   // K = 16+8*half .. +8
  v16bf out;
#pragma unroll
  for (int i = 0; i < 8; ++i) { out[i] = lo[i]; out[i + 8] = hi[i]; }
  return out;
}

// Same pattern, but source is f32 (normalized softmax scores) -> cvt to bf16.
__device__ __forceinline__ v16bf load_frag_f32_as_bf16(const float* base, int row_stride) {
  int lane = threadIdx.x & 31;
  int r    = lane & 15;
  int half = lane >> 4;
  const float* p = base + (size_t)r * row_stride + half * 8;
  v8f lo = *(const v8f*)(p);
  v8f hi = *(const v8f*)(p + 16);
  v16bf out;
#pragma unroll
  for (int i = 0; i < 8; ++i) { out[i] = (__bf16)lo[i]; out[i + 8] = (__bf16)hi[i]; }
  return out;
}

// ---------------------------------------------------------------------------
// f32 -> bf16 elementwise convert
// ---------------------------------------------------------------------------
__global__ void cvt_f32_bf16(const float* __restrict__ src, __bf16* __restrict__ dst, int n) {
  int i = blockIdx.x * blockDim.x + threadIdx.x;
  if (i < n) dst[i] = (__bf16)src[i];
}

// ---------------------------------------------------------------------------
// GEMM: Y[M,N] = X[M,K] * W[N,K]^T,  M=4096, N=K=1024, bf16 in, f32 acc.
// Block = 128 threads = 4 waves. Each wave computes a 32x64 block
// (2x4 tiles of 16x16): 8 WMMAs per 32-wide k-step, 32 k-steps.
// Grid: (GN/256, GM/32).  MODE is a template parameter so each epilogue is
// straight-line code (no scalar mode dispatch replicated through the unroll).
// MODE 0: Q   -> bf16 [B,H,S,HD], scaled by 1/sqrt(HD)
// MODE 1: K   -> bf16 [B,H,S,HD]
// MODE 2: V   -> bf16 [B,H,HD,S]   (transposed so PV B-operand is row-major)
// MODE 3: out -> f32  [B,S,D] (flat m-major)
// ---------------------------------------------------------------------------
template <int MODE>
__global__ __launch_bounds__(128) void gemm_xwT(const __bf16* __restrict__ X,
                                                const __bf16* __restrict__ W,
                                                void* __restrict__ Y) {
  int wave = threadIdx.x >> 5;
  int m0 = blockIdx.y * 32;
  int n0 = blockIdx.x * 256 + wave * 64;

  v8f acc[2][4] = {};
  for (int k0 = 0; k0 < GK; k0 += 32) {
    v16bf a0 = load_frag_bf16(X + (size_t)m0 * GK + k0, GK);
    v16bf a1 = load_frag_bf16(X + (size_t)(m0 + 16) * GK + k0, GK);
    v16bf bfrag[4];
#pragma unroll
    for (int j = 0; j < 4; ++j)
      bfrag[j] = load_frag_bf16(W + (size_t)(n0 + j * 16) * GK + k0, GK);
    if (k0 + 32 < GK) {
      // gfx1250 global_prefetch_b8 for the next k-step (HBM-cold first pass)
      __builtin_prefetch(X + (size_t)m0 * GK + k0 + 32, 0, 1);
      __builtin_prefetch(W + (size_t)n0 * GK + k0 + 32, 0, 1);
    }
#pragma unroll
    for (int j = 0; j < 4; ++j) {
      acc[0][j] = wmma_bf16(a0, bfrag[j], acc[0][j]);
      acc[1][j] = wmma_bf16(a1, bfrag[j], acc[1][j]);
    }
  }

  // C/D layout: lane holds column n = lane%16, rows m = 8*(lane/16)+v.
  int lane = threadIdx.x & 31;
  int nl   = lane & 15;
  int half = lane >> 4;
#pragma unroll
  for (int i = 0; i < 2; ++i)
#pragma unroll
    for (int j = 0; j < 4; ++j)
#pragma unroll
      for (int v = 0; v < 8; ++v) {
        int m = m0 + i * 16 + 8 * half + v;
        int n = n0 + j * 16 + nl;
        float val = acc[i][j][v];
        if constexpr (MODE == 3) {
          ((float*)Y)[(size_t)m * GN + n] = val;
        } else {
          int b = m >> 11, s = m & (S_ - 1);
          int h = n >> 6,  hd = n & (HD_ - 1);
          __bf16* Yb = (__bf16*)Y;
          if constexpr (MODE == 0)
            Yb[(((size_t)(b * H_ + h) * S_) + s) * HD_ + hd] = (__bf16)(val * 0.125f);
          else if constexpr (MODE == 1)
            Yb[(((size_t)(b * H_ + h) * S_) + s) * HD_ + hd] = (__bf16)val;
          else
            Yb[(((size_t)(b * H_ + h) * HD_) + hd) * S_ + s] = (__bf16)val;
        }
      }
}

// ---------------------------------------------------------------------------
// Attention: one workgroup (4 waves) per (b, h, 16-row q band).
// Phase 1: raw scores = Q*K^T (scale folded into Q) + analytic causal mask,
//          stored f32 into d_out scores region (each wave: 32 col-tiles,
//          2 WMMAs each).
// Phase 2: in-place softmax over the 16x2048 band (LDS tree reduction,
//          8 threads per row, v8f-vectorized passes).
// Phase 3: out = P * V via WMMA (A frags cvt'd f32->bf16 on the fly,
//          B frags from transposed V), bf16 -> attn buffer [B,S,D].
// ---------------------------------------------------------------------------
__global__ __launch_bounds__(128) void attn_softmax_pv(const __bf16* __restrict__ Q,
                                                       const __bf16* __restrict__ Km,
                                                       const __bf16* __restrict__ Vt,
                                                       float* __restrict__ scores,
                                                       __bf16* __restrict__ attn) {
  __shared__ float red_max[16][8];
  __shared__ float red_sum[16][8];

  int wg   = blockIdx.x;
  int band = wg & 127;
  int h    = (wg >> 7) & (H_ - 1);
  int b    = wg >> 11;
  int bh   = b * H_ + h;
  int q0   = band * 16;

  int wave = threadIdx.x >> 5;
  int lane = threadIdx.x & 31;
  int nl   = lane & 15;
  int half = lane >> 4;

  const __bf16* Qband = Q + ((size_t)bh * S_ + q0) * HD_;
  float*        sband = scores + ((size_t)bh * S_ + q0) * S_;

  // ---- Phase 1: raw masked scores ----
  v16bf aq0 = load_frag_bf16(Qband,      HD_);   // K = 0..31 of head dim
  v16bf aq1 = load_frag_bf16(Qband + 32, HD_);   // K = 32..63

  for (int t = wave * 32; t < wave * 32 + 32; ++t) {
    int n0 = t * 16;
    const __bf16* Kt = Km + ((size_t)bh * S_ + n0) * HD_;
    v16bf bk0 = load_frag_bf16(Kt,      HD_);
    v16bf bk1 = load_frag_bf16(Kt + 32, HD_);
    v8f acc = {};
    acc = wmma_bf16(aq0, bk0, acc);
    acc = wmma_bf16(aq1, bk1, acc);
    int col = n0 + nl;
#pragma unroll
    for (int v = 0; v < 8; ++v) {
      int row = q0 + 8 * half + v;
      float x = acc[v] + (col > row ? -1.0e9f : 0.0f);   // causal additive mask
      sband[(size_t)(8 * half + v) * S_ + col] = x;
    }
  }
  __threadfence_block();
  __syncthreads();

  // ---- Phase 2: softmax over each of the 16 rows (8 threads/row) ----
  {
    int rl  = threadIdx.x >> 3;   // row 0..15
    int seg = threadIdx.x & 7;    // 256-wide segment
    float* row = sband + (size_t)rl * S_ + seg * 256;

    float mloc = -3.0e38f;
    for (int i = 0; i < 256; i += 8) {
      v8f x = *(const v8f*)(row + i);
#pragma unroll
      for (int j = 0; j < 8; ++j) mloc = fmaxf(mloc, x[j]);
    }
    red_max[rl][seg] = mloc;
    __syncthreads();
    float rmax = red_max[rl][0];
#pragma unroll
    for (int j = 1; j < 8; ++j) rmax = fmaxf(rmax, red_max[rl][j]);

    float ssum = 0.0f;
    for (int i = 0; i < 256; i += 8) {
      v8f x = *(const v8f*)(row + i);
#pragma unroll
      for (int j = 0; j < 8; ++j) ssum += __expf(x[j] - rmax);
    }
    red_sum[rl][seg] = ssum;
    __syncthreads();
    float rsum = 0.0f;
#pragma unroll
    for (int j = 0; j < 8; ++j) rsum += red_sum[rl][j];
    float inv = 1.0f / rsum;

    for (int i = 0; i < 256; i += 8) {
      v8f x = *(const v8f*)(row + i);
#pragma unroll
      for (int j = 0; j < 8; ++j) x[j] = __expf(x[j] - rmax) * inv;
      *(v8f*)(row + i) = x;
    }
  }
  __threadfence_block();
  __syncthreads();

  // ---- Phase 3: PV. Each wave owns one 16-wide slice of head dim. ----
  {
    const __bf16* Vrow = Vt + ((size_t)bh * HD_ + wave * 16) * S_;  // [hd][s]
    v8f acc = {};
    for (int kt = 0; kt < S_ / 32; ++kt) {
      v16bf ap = load_frag_f32_as_bf16(sband + kt * 32, S_);
      v16bf bv = load_frag_bf16(Vrow + kt * 32, S_);
      acc = wmma_bf16(ap, bv, acc);
    }
#pragma unroll
    for (int v = 0; v < 8; ++v) {
      int srow = q0 + 8 * half + v;
      int dcol = h * HD_ + wave * 16 + nl;
      attn[((size_t)(b * S_ + srow)) * D_ + dcol] = (__bf16)acc[v];
    }
  }
}

// ---------------------------------------------------------------------------
// Launch
// ---------------------------------------------------------------------------
extern "C" void kernel_launch(void* const* d_in, const int* in_sizes, int n_in,
                              void* d_out, int out_size, void* d_ws, size_t ws_size,
                              hipStream_t stream) {
  (void)in_sizes; (void)n_in; (void)out_size; (void)ws_size;

  const float* x_q = (const float*)d_in[0];
  const float* x_k = (const float*)d_in[1];
  const float* x_v = (const float*)d_in[2];
  // d_in[3] freqs_complex, d_in[4] mask, d_in[5] rope: unused (mask applied analytically)
  const float* wq = (const float*)d_in[6];
  const float* wk = (const float*)d_in[7];
  const float* wv = (const float*)d_in[8];
  const float* wo = (const float*)d_in[9];

  float* out_f32 = (float*)d_out;                              // [B,S,D]
  float* scores  = (float*)d_out + (size_t)B_ * S_ * D_;       // [B,H,S,S]

  // Workspace layout (bf16), 64 MB total
  char* ws = (char*)d_ws;
  const size_t XN = (size_t)GM * GK;             // 4M elems
  const size_t WN = (size_t)GN * GK;             // 1M elems
  const size_t QN = (size_t)B_ * H_ * S_ * HD_;  // 4M elems
  __bf16* Xq  = (__bf16*)ws;                ws += XN * 2;
  __bf16* Xk  = (__bf16*)ws;                ws += XN * 2;
  __bf16* Xv  = (__bf16*)ws;                ws += XN * 2;
  __bf16* Wq  = (__bf16*)ws;                ws += WN * 2;
  __bf16* Wk  = (__bf16*)ws;                ws += WN * 2;
  __bf16* Wv  = (__bf16*)ws;                ws += WN * 2;
  __bf16* Wo  = (__bf16*)ws;                ws += WN * 2;
  __bf16* Qb  = (__bf16*)ws;                ws += QN * 2;
  __bf16* Kb  = (__bf16*)ws;                ws += QN * 2;
  __bf16* Vtb = (__bf16*)ws;                ws += QN * 2;
  __bf16* At  = (__bf16*)ws;                ws += XN * 2;

  // 1) precision convert
  {
    int nX = (int)XN, nW = (int)WN;
    dim3 blk(256);
    cvt_f32_bf16<<<(nX + 255) / 256, blk, 0, stream>>>(x_q, Xq, nX);
    cvt_f32_bf16<<<(nX + 255) / 256, blk, 0, stream>>>(x_k, Xk, nX);
    cvt_f32_bf16<<<(nX + 255) / 256, blk, 0, stream>>>(x_v, Xv, nX);
    cvt_f32_bf16<<<(nW + 255) / 256, blk, 0, stream>>>(wq, Wq, nW);
    cvt_f32_bf16<<<(nW + 255) / 256, blk, 0, stream>>>(wk, Wk, nW);
    cvt_f32_bf16<<<(nW + 255) / 256, blk, 0, stream>>>(wv, Wv, nW);
    cvt_f32_bf16<<<(nW + 255) / 256, blk, 0, stream>>>(wo, Wo, nW);
  }

  // 2) QKV projections (WMMA bf16). Scale 1/sqrt(64) folded into Q; V transposed.
  {
    dim3 grid(GN / 256, GM / 32), blk(128);
    gemm_xwT<0><<<grid, blk, 0, stream>>>(Xq, Wq, (void*)Qb);
    gemm_xwT<1><<<grid, blk, 0, stream>>>(Xk, Wk, (void*)Kb);
    gemm_xwT<2><<<grid, blk, 0, stream>>>(Xv, Wv, (void*)Vtb);
  }

  // 3) attention: scores + softmax + PV
  {
    dim3 grid(B_ * H_ * (S_ / 16)), blk(128);
    attn_softmax_pv<<<grid, blk, 0, stream>>>(Qb, Kb, Vtb, scores, At);
  }

  // 4) output projection -> f32 out
  {
    dim3 grid(GN / 256, GM / 32), blk(128);
    gemm_xwT<3><<<grid, blk, 0, stream>>>(At, Wo, (void*)out_f32);
  }
}
